// CatalyticDiffusionModel_50070728736887
// MI455X (gfx1250) — matrix operand order repaired and checked by reading, not compiled, not run on hardware
//
#include <hip/hip_runtime.h>
#include <cstddef>
#include <cstdint>

#define HID 256
#define NLAYERS 6
#define EIN_STRIDE 520   // bf16 halves per row (512 + 8 pad)
#define MSTR_H 264       // bf16 halves per row (256 + 8 pad)
#define MSTR_F 260       // floats per row (256 + 4 pad)

typedef __attribute__((ext_vector_type(16))) __bf16 v16bf;
typedef __attribute__((ext_vector_type(8)))  float  v8f;
typedef __attribute__((ext_vector_type(4)))  unsigned int u32x4;

union Frag { v16bf v; u32x4 q[2]; };

__device__ __forceinline__ unsigned short f2bf(float f){
  unsigned int u = __float_as_uint(f);
  unsigned int r = u + 0x7FFFu + ((u >> 16) & 1u);
  return (unsigned short)(r >> 16);
}
__device__ __forceinline__ float siluf(float x){ return x / (1.f + __expf(-x)); }

__device__ __forceinline__ float block_sum256(float v, float* red, int k){
  red[k] = v; __syncthreads();
  for (int off = 128; off > 0; off >>= 1){
    if (k < off) red[k] += red[k + off];
    __syncthreads();
  }
  float s = red[0]; __syncthreads();
  return s;
}

// ---------------------------------------------------------------------------
// Pack a row-major f32 weight [K][256] into WMMA B-fragment bf16 layout:
//  out[((nt*KT + kc)*32 + lane)*16 + h] = W[kc*32 + (lane>>4)*8 + (h&7) + (h>>3)*16][nt*16 + (lane&15)]
// ---------------------------------------------------------------------------
__global__ void pack_w(const float* __restrict__ W, unsigned short* __restrict__ out,
                       int K, int Ncols){
  int idx = blockIdx.x * 256 + threadIdx.x;
  int total = K * Ncols;
  if (idx >= total) return;
  int h    = idx & 15;
  int lane = (idx >> 4) & 31;
  int kt   = K >> 5;
  int rest = idx >> 9;
  int kc   = rest % kt;
  int nt   = rest / kt;
  int k = kc * 32 + ((lane >> 4) << 3) + (h & 7) + ((h >> 3) << 4);
  int n = nt * 16 + (lane & 15);
  out[idx] = f2bf(W[(size_t)k * Ncols + n]);
}

// ---------------------------------------------------------------------------
// Front-end: time embedding + condition encoder + projections (B<=4, tiny)
// ---------------------------------------------------------------------------
struct FrontArgs {
  const int* t;
  const float *anc, *dstc, *crdc;
  const float *tW1,*tb1,*tW2,*tb2;
  const float *aW1,*ab1,*aW2,*ab2;
  const float *dW1,*db1,*dW2,*db2;
  const float *cW1,*cb1,*cW2,*cb2;
  const float *fW1,*fb1,*flg,*flb,*fW2,*fb2;
  const float *tpW,*tpb,*cpW,*cpb;
  float* tc;
  int B;
};

__global__ __launch_bounds__(256) void frontend_kernel(FrontArgs A){
  __shared__ float sA[4*256], sD[4*256], sC[4*256], sT[4*256];
  __shared__ float tmp[256], tmp2[256], red[256];
  const int k = threadIdx.x;
  int B = A.B; if (B > 4) B = 4;

  // time embedding
  for (int b = 0; b < B; ++b){
    float tv = (float)A.t[b];
    float te;
    if (k < 128) te = sinf(tv * __expf(-logf(10000.f) * (float)k / 127.f));
    else         te = cosf(tv * __expf(-logf(10000.f) * (float)(k-128) / 127.f));
    tmp[k] = te; __syncthreads();
    float acc = A.tb1[k];
    for (int j = 0; j < 256; ++j) acc += tmp[j] * A.tW1[j*256 + k];
    tmp2[k] = siluf(acc); __syncthreads();
    acc = A.tb2[k];
    for (int j = 0; j < 256; ++j) acc += tmp2[j] * A.tW2[j*256 + k];
    sT[b*256 + k] = acc; __syncthreads();
  }

  // anchor encoder: (16->256 silu, 256->256), mean over 8
  float accA[4] = {0.f,0.f,0.f,0.f};
  for (int b = 0; b < B; ++b)
    for (int i = 0; i < 8; ++i){
      float acc = A.ab1[k];
      for (int j = 0; j < 16; ++j) acc += A.anc[(b*8 + i)*16 + j] * A.aW1[j*256 + k];
      tmp[k] = siluf(acc); __syncthreads();
      float a2 = 0.f;
      for (int j = 0; j < 256; ++j) a2 += tmp[j] * A.aW2[j*256 + k];
      accA[b] += a2; __syncthreads();
    }
  for (int b = 0; b < B; ++b) sA[b*256 + k] = accA[b] * 0.125f + A.ab2[k];

  // distance encoder: (4->128 silu, 128->256), mean over 4
  float accD[4] = {0.f,0.f,0.f,0.f};
  for (int b = 0; b < B; ++b)
    for (int i = 0; i < 4; ++i){
      if (k < 128){
        float acc = A.db1[k];
        for (int j = 0; j < 4; ++j) acc += A.dstc[(b*4 + i)*4 + j] * A.dW1[j*128 + k];
        tmp[k] = siluf(acc);
      }
      __syncthreads();
      float a2 = 0.f;
      for (int j = 0; j < 128; ++j) a2 += tmp[j] * A.dW2[j*256 + k];
      accD[b] += a2; __syncthreads();
    }
  for (int b = 0; b < B; ++b) sD[b*256 + k] = accD[b] * 0.25f + A.db2[k];

  // coordination encoder: (3->128 silu, 128->256), mean over 4
  float accC[4] = {0.f,0.f,0.f,0.f};
  for (int b = 0; b < B; ++b)
    for (int i = 0; i < 4; ++i){
      if (k < 128){
        float acc = A.cb1[k];
        for (int j = 0; j < 3; ++j) acc += A.crdc[(b*4 + i)*3 + j] * A.cW1[j*128 + k];
        tmp[k] = siluf(acc);
      }
      __syncthreads();
      float a2 = 0.f;
      for (int j = 0; j < 128; ++j) a2 += tmp[j] * A.cW2[j*256 + k];
      accC[b] += a2; __syncthreads();
    }
  for (int b = 0; b < B; ++b) sC[b*256 + k] = accC[b] * 0.25f + A.cb2[k];
  __syncthreads();

  // fusion: linear(768->256) -> LN -> silu -> linear(256->256)
  for (int b = 0; b < B; ++b){
    float acc = A.fb1[k];
    for (int j = 0; j < 256; ++j){
      acc += sA[b*256 + j] * A.fW1[(      j)*256 + k];
      acc += sD[b*256 + j] * A.fW1[(256 + j)*256 + k];
      acc += sC[b*256 + j] * A.fW1[(512 + j)*256 + k];
    }
    float mu  = block_sum256(acc, red, k) * (1.f/256.f);
    float dv  = acc - mu;
    float var = block_sum256(dv*dv, red, k) * (1.f/256.f);
    float ln  = A.flg[k] * dv * rsqrtf(var + 1e-5f) + A.flb[k];
    tmp[k] = siluf(ln); __syncthreads();
    float acc2 = A.fb2[k];
    for (int j = 0; j < 256; ++j) acc2 += tmp[j] * A.fW2[j*256 + k];
    sA[b*256 + k] = acc2;  // cond_emb (sA reuse)
    __syncthreads();
  }

  // projections: tc = t_emb@tpW + tpb + cond_emb@cpW + cpb
  for (int b = 0; b < B; ++b){
    float tp = A.tpb[k], cp = A.cpb[k];
    for (int j = 0; j < 256; ++j){
      tp += sT[b*256 + j] * A.tpW[j*256 + k];
      cp += sA[b*256 + j] * A.cpW[j*256 + k];
    }
    A.tc[b*256 + k] = tp + cp;
  }
}

// ---------------------------------------------------------------------------
// h init: h = embed[atom_types] + tc[node/nps]; also bf16 mirror; copy x
// ---------------------------------------------------------------------------
__global__ void init_h(const int* __restrict__ at, const float* __restrict__ embed,
                       const float* __restrict__ tc, const float* __restrict__ x_in,
                       float* __restrict__ h, unsigned short* __restrict__ hbf,
                       float* __restrict__ x0, int N, int nps){
  int i = blockIdx.x * 256 + threadIdx.x;
  if (i < N * HID){
    int n = i >> 8, k = i & 255;
    float v = embed[(size_t)at[n] * HID + k] + tc[(n / nps) * HID + k];
    h[i] = v; hbf[i] = f2bf(v);
  }
  if (i < N * 3) x0[i] = x_in[i];
}

__global__ void zero_copy(float* __restrict__ m_agg, float* __restrict__ xdst,
                          const float* __restrict__ xsrc, int nM, int nX){
  int i = blockIdx.x * 256 + threadIdx.x;
  if (i < nM) m_agg[i] = 0.f;
  if (i < nX) xdst[i] = xsrc[i];
}

// ---------------------------------------------------------------------------
// Edge kernel: 16 edges/block, 16 waves, wave = N-tile. All GEMMs via WMMA bf16.
// ---------------------------------------------------------------------------
__global__ __launch_bounds__(512) void egnn_edge(
    const int* __restrict__ rowIdx, const int* __restrict__ colIdx,
    const float* __restrict__ x_cur, float* __restrict__ x_next,
    const unsigned short* __restrict__ hbf, float* __restrict__ m_agg,
    const unsigned short* __restrict__ eW1p, const unsigned short* __restrict__ eW2p,
    const unsigned short* __restrict__ cW1p,
    const float* __restrict__ eb1, const float* __restrict__ w1dist,
    const float* __restrict__ eb2, const float* __restrict__ aWv,
    const float* __restrict__ abv, const float* __restrict__ cb1v,
    const float* __restrict__ cW2v, int E){
  __shared__ unsigned short s_ein[16 * EIN_STRIDE];
  __shared__ unsigned short s_m1 [16 * MSTR_H];
  __shared__ unsigned short s_mbf[16 * MSTR_H];
  __shared__ float          s_mf [16 * MSTR_F];
  __shared__ int   s_nr[16], s_nc[16];
  __shared__ float s_rel[16][3], s_dist[16], s_att[16], s_cwv[16];

  const int tid  = threadIdx.x;
  const int lane = tid & 31;
  const int wave = tid >> 5;       // = ntile = edge slot for staging
  const int e0   = blockIdx.x * 16;

  if (tid < 16){
    int e = e0 + tid; if (e >= E) e = E - 1;
    int r = rowIdx[e], c = colIdx[e];
    s_nr[tid] = r; s_nc[tid] = c;
    float dx = x_cur[r*3+0] - x_cur[c*3+0];
    float dy = x_cur[r*3+1] - x_cur[c*3+1];
    float dz = x_cur[r*3+2] - x_cur[c*3+2];
    s_rel[tid][0] = dx; s_rel[tid][1] = dy; s_rel[tid][2] = dz;
    s_dist[tid] = sqrtf(dx*dx + dy*dy + dz*dz);
  }
  __syncthreads();

  { // stage ein[16][512] = concat(hbf[row], hbf[col]); each lane copies 16 halves
    const unsigned short* src = (lane < 16)
        ? hbf + (size_t)s_nr[wave] * HID + lane * 16
        : hbf + (size_t)s_nc[wave] * HID + (lane - 16) * 16;
    unsigned short* dst = &s_ein[wave * EIN_STRIDE + lane * 16];
    *(u32x4*)dst       = *(const u32x4*)src;
    *(u32x4*)(dst + 8) = *(const u32x4*)(src + 8);
  }
  __syncthreads();

  const int ntile = wave;
  const int mA    = lane & 15;
  const int koff  = (lane >> 4) * 8;
  const int laneN = ntile * 16 + (lane & 15);
  const int mBase = (lane >> 4) * 8;

  // ---- GEMM1: ein[16x512] @ eW1 -> m1 (bias + dist rank-1 + silu) ----
  {
    v8f acc = {};
    for (int kc = 0; kc < 16; ++kc){
      Frag a, b;
      const unsigned short* ap = &s_ein[mA * EIN_STRIDE + kc*32 + koff];
      a.q[0] = *(const u32x4*)ap;
      a.q[1] = *(const u32x4*)(ap + 16);
      const u32x4* wp = (const u32x4*)(eW1p + (((size_t)ntile*16 + kc)*32 + lane)*16);
      b.q[0] = wp[0]; b.q[1] = wp[1];
      acc = __builtin_amdgcn_wmma_f32_16x16x32_bf16(false, a.v, false, b.v,
                                                    (short)0, acc, false, false);
    }
    float bias = eb1[laneN], wd = w1dist[laneN];
    for (int r = 0; r < 8; ++r){
      int m = mBase + r;
      float val = siluf(acc[r] + bias + s_dist[m] * wd);
      s_m1[m * MSTR_H + laneN] = f2bf(val);
    }
  }
  __syncthreads();

  // ---- GEMM2: m1 @ eW2 + eb2, silu -> m (f32 + bf16 copies) ----
  {
    v8f acc = {};
    for (int kc = 0; kc < 8; ++kc){
      Frag a, b;
      const unsigned short* ap = &s_m1[mA * MSTR_H + kc*32 + koff];
      a.q[0] = *(const u32x4*)ap;
      a.q[1] = *(const u32x4*)(ap + 16);
      const u32x4* wp = (const u32x4*)(eW2p + (((size_t)ntile*8 + kc)*32 + lane)*16);
      b.q[0] = wp[0]; b.q[1] = wp[1];
      acc = __builtin_amdgcn_wmma_f32_16x16x32_bf16(false, a.v, false, b.v,
                                                    (short)0, acc, false, false);
    }
    float bias = eb2[laneN];
    for (int r = 0; r < 8; ++r){
      int m = mBase + r;
      float val = siluf(acc[r] + bias);
      s_mf [m * MSTR_F + laneN] = val;
      s_mbf[m * MSTR_H + laneN] = f2bf(val);
    }
  }
  __syncthreads();

  // ---- attention: sigmoid(m . aW + ab) ----
  {
    float p = 0.f;
    for (int j = 0; j < 8; ++j){
      int col = lane + j*32;
      p += s_mf[wave * MSTR_F + col] * aWv[col];
    }
    for (int off = 16; off > 0; off >>= 1) p += __shfl_xor(p, off, 32);
    if (lane == 0) s_att[wave] = 1.f / (1.f + __expf(-(p + abv[0])));
  }
  __syncthreads();

  // ---- scatter att*m into m_agg ----
  if (e0 + wave < E){
    float att = s_att[wave];
    float* dstp = m_agg + (size_t)s_nr[wave] * HID;
    for (int j = 0; j < 8; ++j){
      int col = lane + j*32;
      atomicAdd(dstp + col, att * s_mf[wave * MSTR_F + col]);
    }
  }
  __syncthreads();

  // ---- GEMM3: silu(m @ cW1 + cb1) -> reuse s_mf ----
  {
    v8f acc = {};
    for (int kc = 0; kc < 8; ++kc){
      Frag a, b;
      const unsigned short* ap = &s_mbf[mA * MSTR_H + kc*32 + koff];
      a.q[0] = *(const u32x4*)ap;
      a.q[1] = *(const u32x4*)(ap + 16);
      const u32x4* wp = (const u32x4*)(cW1p + (((size_t)ntile*8 + kc)*32 + lane)*16);
      b.q[0] = wp[0]; b.q[1] = wp[1];
      acc = __builtin_amdgcn_wmma_f32_16x16x32_bf16(false, a.v, false, b.v,
                                                    (short)0, acc, false, false);
    }
    float bias = cb1v[laneN];
    for (int r = 0; r < 8; ++r){
      int m = mBase + r;
      s_mf[m * MSTR_F + laneN] = siluf(acc[r] + bias);
    }
  }
  __syncthreads();

  // ---- cw = (.) @ cW2 ----
  {
    float p = 0.f;
    for (int j = 0; j < 8; ++j){
      int col = lane + j*32;
      p += s_mf[wave * MSTR_F + col] * cW2v[col];
    }
    for (int off = 16; off > 0; off >>= 1) p += __shfl_xor(p, off, 32);
    if (lane == 0) s_cwv[wave] = p;
  }
  __syncthreads();

  // ---- x scatter: x_next[row] += cw * rel/(dist+1e-8) ----
  if (tid < 64){
    int e = tid >> 2, d = tid & 3;
    if (d < 3 && (e0 + e) < E){
      float coef = s_cwv[e] / (s_dist[e] + 1e-8f);
      atomicAdd(x_next + (size_t)s_nr[e]*3 + d, coef * s_rel[e][d]);
    }
  }
}

// ---------------------------------------------------------------------------
// Node kernel: h += MLP(concat(h, m_agg)); rewrite f32 + bf16 h
// ---------------------------------------------------------------------------
__global__ __launch_bounds__(512) void egnn_node(
    float* __restrict__ h, unsigned short* __restrict__ hbf,
    const float* __restrict__ m_agg,
    const unsigned short* __restrict__ nW1p, const unsigned short* __restrict__ nW2p,
    const float* __restrict__ nb1, const float* __restrict__ nb2, int N){
  __shared__ unsigned short s_ein[16 * EIN_STRIDE];
  __shared__ unsigned short s_t1 [16 * MSTR_H];

  const int tid  = threadIdx.x;
  const int lane = tid & 31;
  const int wave = tid >> 5;
  const int n0   = blockIdx.x * 16;

  { // stage concat(h_bf16, bf16(m_agg))
    int node = n0 + wave; if (node >= N) node = N - 1;
    unsigned short* dst = &s_ein[wave * EIN_STRIDE + lane * 16];
    if (lane < 16){
      const unsigned short* src = hbf + (size_t)node * HID + lane * 16;
      *(u32x4*)dst       = *(const u32x4*)src;
      *(u32x4*)(dst + 8) = *(const u32x4*)(src + 8);
    } else {
      const float* src = m_agg + (size_t)node * HID + (lane - 16) * 16;
      for (int j = 0; j < 16; ++j) dst[j] = f2bf(src[j]);
    }
  }
  __syncthreads();

  const int ntile = wave;
  const int mA    = lane & 15;
  const int koff  = (lane >> 4) * 8;
  const int laneN = ntile * 16 + (lane & 15);
  const int mBase = (lane >> 4) * 8;

  { // GEMM1: K=512 -> silu
    v8f acc = {};
    for (int kc = 0; kc < 16; ++kc){
      Frag a, b;
      const unsigned short* ap = &s_ein[mA * EIN_STRIDE + kc*32 + koff];
      a.q[0] = *(const u32x4*)ap;
      a.q[1] = *(const u32x4*)(ap + 16);
      const u32x4* wp = (const u32x4*)(nW1p + (((size_t)ntile*16 + kc)*32 + lane)*16);
      b.q[0] = wp[0]; b.q[1] = wp[1];
      acc = __builtin_amdgcn_wmma_f32_16x16x32_bf16(false, a.v, false, b.v,
                                                    (short)0, acc, false, false);
    }
    float bias = nb1[laneN];
    for (int r = 0; r < 8; ++r){
      int m = mBase + r;
      s_t1[m * MSTR_H + laneN] = f2bf(siluf(acc[r] + bias));
    }
  }
  __syncthreads();

  { // GEMM2: K=256, residual add, rewrite h/hbf
    v8f acc = {};
    for (int kc = 0; kc < 8; ++kc){
      Frag a, b;
      const unsigned short* ap = &s_t1[mA * MSTR_H + kc*32 + koff];
      a.q[0] = *(const u32x4*)ap;
      a.q[1] = *(const u32x4*)(ap + 16);
      const u32x4* wp = (const u32x4*)(nW2p + (((size_t)ntile*8 + kc)*32 + lane)*16);
      b.q[0] = wp[0]; b.q[1] = wp[1];
      acc = __builtin_amdgcn_wmma_f32_16x16x32_bf16(false, a.v, false, b.v,
                                                    (short)0, acc, false, false);
    }
    float bias = nb2[laneN];
    for (int r = 0; r < 8; ++r){
      int m = mBase + r;
      int node = n0 + m;
      if (node < N){
        size_t idx = (size_t)node * HID + laneN;
        float hv = h[idx] + acc[r] + bias;
        h[idx] = hv; hbf[idx] = f2bf(hv);
      }
    }
  }
}

// ---------------------------------------------------------------------------
// Final: layernorm + head (256->100); emit logits then x
// ---------------------------------------------------------------------------
__global__ __launch_bounds__(256) void final_head(
    const float* __restrict__ h, const float* __restrict__ g, const float* __restrict__ bta,
    const float* __restrict__ headW, const float* __restrict__ headb,
    const float* __restrict__ xf, float* __restrict__ out, int N){
  __shared__ float hn[256], red[256];
  const int node = blockIdx.x;
  const int k = threadIdx.x;
  float v   = h[(size_t)node * HID + k];
  float mu  = block_sum256(v, red, k) * (1.f/256.f);
  float dv  = v - mu;
  float var = block_sum256(dv*dv, red, k) * (1.f/256.f);
  hn[k] = g[k] * dv * rsqrtf(var + 1e-5f) + bta[k];
  __syncthreads();
  if (k < 100){
    float acc = headb[k];
    for (int j = 0; j < 256; ++j) acc += hn[j] * headW[j*100 + k];
    out[(size_t)node * 100 + k] = acc;
  }
  if (k < 3) out[(size_t)N * 100 + (size_t)node * 3 + k] = xf[(size_t)node * 3 + k];
}

// ---------------------------------------------------------------------------
// Host launcher
// ---------------------------------------------------------------------------
extern "C" void kernel_launch(void* const* d_in, const int* in_sizes, int n_in,
                              void* d_out, int out_size, void* d_ws, size_t ws_size,
                              hipStream_t stream) {
  (void)n_in; (void)out_size; (void)ws_size;
  const int N = in_sizes[0];
  const int E = in_sizes[2] / 2;
  const int B = in_sizes[3];
  const int nps = N / B;

  const int*   atypes = (const int*)  d_in[0];
  const float* x_in   = (const float*)d_in[1];
  const int*   eidx   = (const int*)  d_in[2];
  const float* embed  = (const float*)d_in[7];

  // layers
  const float* LEW1 = (const float*)d_in[34];
  const float* LEB1 = (const float*)d_in[35];
  const float* LEW2 = (const float*)d_in[36];
  const float* LEB2 = (const float*)d_in[37];
  const float* LNW1 = (const float*)d_in[38];
  const float* LNB1 = (const float*)d_in[39];
  const float* LNW2 = (const float*)d_in[40];
  const float* LNB2 = (const float*)d_in[41];
  const float* LCW1 = (const float*)d_in[42];
  const float* LCB1 = (const float*)d_in[43];
  const float* LCW2 = (const float*)d_in[44];
  const float* LAW  = (const float*)d_in[45];
  const float* LAB  = (const float*)d_in[46];
  const float* NORMG = (const float*)d_in[47];
  const float* NORMB = (const float*)d_in[48];
  const float* HEADW = (const float*)d_in[49];
  const float* HEADB = (const float*)d_in[50];

  // workspace bump allocator
  char* wp = (char*)d_ws;
  auto alloc = [&](size_t bytes) -> void* {
    void* p = (void*)wp;
    wp += (bytes + 255) & ~(size_t)255;
    return p;
  };
  float*          tc    = (float*)alloc((size_t)(B > 4 ? B : 4) * HID * 4);
  float*          h     = (float*)alloc((size_t)N * HID * 4);
  unsigned short* hbf   = (unsigned short*)alloc((size_t)N * HID * 2);
  float*          m_agg = (float*)alloc((size_t)N * HID * 4);
  float*          x0    = (float*)alloc((size_t)N * 3 * 4);
  float*          x1    = (float*)alloc((size_t)N * 3 * 4);
  unsigned short* eW1p  = (unsigned short*)alloc((size_t)NLAYERS * 512 * 256 * 2);
  unsigned short* eW2p  = (unsigned short*)alloc((size_t)NLAYERS * 256 * 256 * 2);
  unsigned short* nW1p  = (unsigned short*)alloc((size_t)NLAYERS * 512 * 256 * 2);
  unsigned short* nW2p  = (unsigned short*)alloc((size_t)NLAYERS * 256 * 256 * 2);
  unsigned short* cW1p  = (unsigned short*)alloc((size_t)NLAYERS * 256 * 256 * 2);

  // pack weights -> bf16 WMMA B-fragments
  const int g512 = (512 * 256 + 255) / 256;
  const int g256 = (256 * 256 + 255) / 256;
  for (int l = 0; l < NLAYERS; ++l){
    pack_w<<<g512, 256, 0, stream>>>(LEW1 + (size_t)l*513*256, eW1p + (size_t)l*512*256, 512, 256);
    pack_w<<<g256, 256, 0, stream>>>(LEW2 + (size_t)l*256*256, eW2p + (size_t)l*256*256, 256, 256);
    pack_w<<<g512, 256, 0, stream>>>(LNW1 + (size_t)l*512*256, nW1p + (size_t)l*512*256, 512, 256);
    pack_w<<<g256, 256, 0, stream>>>(LNW2 + (size_t)l*256*256, nW2p + (size_t)l*256*256, 256, 256);
    pack_w<<<g256, 256, 0, stream>>>(LCW1 + (size_t)l*256*256, cW1p + (size_t)l*256*256, 256, 256);
  }

  // front end
  FrontArgs fa;
  fa.t    = (const int*)  d_in[3];
  fa.anc  = (const float*)d_in[4];
  fa.dstc = (const float*)d_in[5];
  fa.crdc = (const float*)d_in[6];
  fa.tW1 = (const float*)d_in[8];  fa.tb1 = (const float*)d_in[9];
  fa.tW2 = (const float*)d_in[10]; fa.tb2 = (const float*)d_in[11];
  fa.aW1 = (const float*)d_in[12]; fa.ab1 = (const float*)d_in[13];
  fa.aW2 = (const float*)d_in[14]; fa.ab2 = (const float*)d_in[15];
  fa.dW1 = (const float*)d_in[16]; fa.db1 = (const float*)d_in[17];
  fa.dW2 = (const float*)d_in[18]; fa.db2 = (const float*)d_in[19];
  fa.cW1 = (const float*)d_in[20]; fa.cb1 = (const float*)d_in[21];
  fa.cW2 = (const float*)d_in[22]; fa.cb2 = (const float*)d_in[23];
  fa.fW1 = (const float*)d_in[24]; fa.fb1 = (const float*)d_in[25];
  fa.flg = (const float*)d_in[26]; fa.flb = (const float*)d_in[27];
  fa.fW2 = (const float*)d_in[28]; fa.fb2 = (const float*)d_in[29];
  fa.tpW = (const float*)d_in[30]; fa.tpb = (const float*)d_in[31];
  fa.cpW = (const float*)d_in[32]; fa.cpb = (const float*)d_in[33];
  fa.tc = tc; fa.B = B;
  frontend_kernel<<<1, 256, 0, stream>>>(fa);

  const int gNH = (N * HID + 255) / 256;
  init_h<<<gNH, 256, 0, stream>>>(atypes, embed, tc, x_in, h, hbf, x0, N, nps);

  float* xc = x0; float* xn = x1;
  const int gridE = (E + 15) / 16;
  const int gridN = (N + 15) / 16;
  for (int l = 0; l < NLAYERS; ++l){
    zero_copy<<<gNH, 256, 0, stream>>>(m_agg, xn, xc, N * HID, N * 3);
    egnn_edge<<<gridE, 512, 0, stream>>>(
        eidx, eidx + E, xc, xn, hbf, m_agg,
        eW1p + (size_t)l*512*256, eW2p + (size_t)l*256*256, cW1p + (size_t)l*256*256,
        LEB1 + (size_t)l*256,
        LEW1 + (size_t)l*513*256 + (size_t)512*256,   // dist row of eW1
        LEB2 + (size_t)l*256,
        LAW  + (size_t)l*256, LAB + l,
        LCB1 + (size_t)l*256, LCW2 + (size_t)l*256, E);
    egnn_node<<<gridN, 512, 0, stream>>>(
        h, hbf, m_agg,
        nW1p + (size_t)l*512*256, nW2p + (size_t)l*256*256,
        LNB1 + (size_t)l*256, LNB2 + (size_t)l*256, N);
    float* t = xc; xc = xn; xn = t;
  }

  final_head<<<N, 256, 0, stream>>>(h, NORMG, NORMB, HEADW, HEADB, xc, (float*)d_out, N);
}
